// edgeAcct_48034914239042
// MI455X (gfx1250) — matrix operand order repaired and checked by reading, not compiled, not run on hardware
//
#include <hip/hip_runtime.h>

typedef __attribute__((ext_vector_type(16))) _Float16 v16h;
typedef __attribute__((ext_vector_type(8)))  _Float16 v8h;
typedef __attribute__((ext_vector_type(8)))  float    v8f;

#define BATCH 8
#define NPIX  4096      // 64*64
#define DDIM  16
#define NTILES 256      // NPIX/16

// ---------------------------------------------------------------------------
// Kernel A: per-pixel QKV projection.
//  Qt, Kt : (B, N, 16) f16   -- n-major, feeds WMMA A (Q) and B (K) fragments
//  Vh     : (B, 16, N) f16   -- d-major, feeds WMMA A fragment of 2nd GEMM
//           (rescaled in place by 1/Z in kernel B)
// ---------------------------------------------------------------------------
__global__ void ea_qkv_kernel(const float* __restrict__ x, const float* __restrict__ e,
                              const float* __restrict__ Wq, const float* __restrict__ bq,
                              const float* __restrict__ Wk, const float* __restrict__ bk,
                              const float* __restrict__ Wv, const float* __restrict__ bv,
                              _Float16* __restrict__ Qt, _Float16* __restrict__ Kt,
                              _Float16* __restrict__ Vh) {
    int idx = blockIdx.x * blockDim.x + threadIdx.x;      // b*N + n
    if (idx >= BATCH * NPIX) return;
    int b = idx / NPIX, n = idx % NPIX;
    float x0 = x[(b * 2 + 0) * NPIX + n];
    float x1 = x[(b * 2 + 1) * NPIX + n];
    float xn = x0 * x0 + x1 * x1;                         // channel 0 of x1
    float ev = e[b * NPIX + n];                           // channel 1 of x1
#pragma unroll
    for (int d = 0; d < DDIM; ++d) {
        float q = Wq[d * 2] * xn + Wq[d * 2 + 1] * ev + bq[d];
        float k = Wk[d * 2] * xn + Wk[d * 2 + 1] * ev + bk[d];
        float v = Wv[d * 2] * x0 + Wv[d * 2 + 1] * x1 + bv[d];
        Qt[(size_t)idx * DDIM + d] = (_Float16)q;
        Kt[(size_t)idx * DDIM + d] = (_Float16)k;
        Vh[((size_t)b * DDIM + d) * NPIX + n] = (_Float16)v;
    }
}

// ---------------------------------------------------------------------------
// Kernel B: per n-tile softmax statistics of S = Q^T K, two WMMA sweeps:
//   sweep 1: rowmax only (v_max per element, no transcendentals)
//   sweep 2: recompute S (L1-hit loads, WMMA nearly free), sum exp(S-max)
//            -> exactly 1 v_exp per S element, no rescale chains.
// B fragments are loaded UNCONDITIONALLY on all 32 lanes: lanes 16-31 feed
// K=16..31, which multiplies the zero upper half of the A fragment, so any
// finite payload is harmless (0 x finite = 0). No EXEC juggling per tile.
// Then rescale V in place by 1/Z. One wave owns one 16-row n-tile.
// ---------------------------------------------------------------------------
__global__ void ea_rowstats_kernel(const _Float16* __restrict__ Qt,
                                   const _Float16* __restrict__ Kt,
                                   _Float16* __restrict__ Vh,
                                   float* __restrict__ rowmax) {
    int tile = blockIdx.x;                 // b*256 + nt
    int b  = tile >> 8;
    int n0 = (tile & 255) << 4;
    int l  = threadIdx.x;                  // 0..31
    int g  = l >> 4;                       // lane half
    int lr = l & 15;

    // A fragment: row M = n0+lr ; lane half 0 carries d = 0..15 ; lane half 1 (K 16..31) -> 0
    v16h a = {};
    v16h qa = *(const v16h*)(Qt + (((size_t)b * NPIX + n0 + lr) * DDIM));
#pragma unroll
    for (int h = 0; h < 16; ++h) a[h] = g ? (_Float16)0 : qa[h];

    const _Float16* kbase = Kt + (size_t)b * NPIX * DDIM;

    // ---- sweep 1: row maxima ----
    float rm[8];
#pragma unroll
    for (int r = 0; r < 8; ++r) rm[r] = -1e30f;
    for (int mt = 0; mt < NTILES; ++mt) {
        v16h bm = *(const v16h*)(kbase + (size_t)(mt * 16 + lr) * DDIM);
        v8f c = {};
        c = __builtin_amdgcn_wmma_f32_16x16x32_f16(false, a, false, bm,
                                                   (short)0, c, false, false);
#pragma unroll
        for (int r = 0; r < 8; ++r) rm[r] = fmaxf(rm[r], c[r]);
    }
#pragma unroll
    for (int off = 1; off < 16; off <<= 1)
#pragma unroll
        for (int r = 0; r < 8; ++r)
            rm[r] = fmaxf(rm[r], __shfl_xor(rm[r], off, 32));

    // ---- sweep 2: Z = sum_m exp(S - rowmax) ----
    float rs[8];
#pragma unroll
    for (int r = 0; r < 8; ++r) rs[r] = 0.0f;
    for (int mt = 0; mt < NTILES; ++mt) {
        v16h bm = *(const v16h*)(kbase + (size_t)(mt * 16 + lr) * DDIM);
        v8f c = {};
        c = __builtin_amdgcn_wmma_f32_16x16x32_f16(false, a, false, bm,
                                                   (short)0, c, false, false);
#pragma unroll
        for (int r = 0; r < 8; ++r) rs[r] += __expf(c[r] - rm[r]);
    }
#pragma unroll
    for (int off = 1; off < 16; off <<= 1)
#pragma unroll
        for (int r = 0; r < 8; ++r)
            rs[r] += __shfl_xor(rs[r], off, 32);

    // ---- outputs ----
    if (lr == 0) {   // lane 0 -> rows 0..7, lane 16 -> rows 8..15
#pragma unroll
        for (int r = 0; r < 8; ++r)
            rowmax[(size_t)b * NPIX + n0 + r + 8 * g] = rm[r];
    }

    // Rescale V in place: lane (g,lr) owns d = lr, n = n0 + 8g + h (h=0..7);
    // row h of this half corresponds to rs[h]. One contiguous v8h per lane.
    float inv[8];
#pragma unroll
    for (int h = 0; h < 8; ++h) inv[h] = 1.0f / rs[h];
    _Float16* vp = Vh + ((size_t)b * DDIM + lr) * NPIX + n0 + 8 * g;
    v8h vv = *(const v8h*)vp;
#pragma unroll
    for (int h = 0; h < 8; ++h) vv[h] = (_Float16)((float)vv[h] * inv[h]);
    *(v8h*)vp = vv;
}

// ---------------------------------------------------------------------------
// Kernel D: out[d, m0:m0+16] = sum_n (V[d,n]/Z[n]) * exp(S[n,m] - rowmax[n])
// One wave per (b, m-tile). B fragment of GEMM1 (K tile at m0) is loop
// invariant; per n-tile: WMMA scores -> exp (rowmax fetched as one v8f) ->
// LDS transpose (16B ds_store_b128) -> second WMMA accumulate into the f32
// C fragment = output tile. B fragments loaded unconditionally (see above).
// ---------------------------------------------------------------------------
__global__ void ea_attnout_kernel(const _Float16* __restrict__ Qt,
                                  const _Float16* __restrict__ Kt,
                                  const _Float16* __restrict__ Vh,
                                  const float* __restrict__ rowmax,
                                  float* __restrict__ out) {
    __shared__ __align__(32) _Float16 Pt[256];   // [col m_local][row n_local]
    int tile = blockIdx.x;                 // b*256 + mt
    int b  = tile >> 8;
    int m0 = (tile & 255) << 4;
    int l  = threadIdx.x;
    int g  = l >> 4;
    int lr = l & 15;

    // GEMM1 B fragment (K tile at columns m0..m0+15), loop invariant
    v16h bk = *(const v16h*)(Kt + ((size_t)b * NPIX + m0 + lr) * DDIM);

    v8f acc = {};
    const _Float16* qbase  = Qt + (size_t)b * NPIX * DDIM;
    const _Float16* vbase  = Vh + (size_t)b * DDIM * NPIX;
    const float*    rmbase = rowmax + (size_t)b * NPIX;

    for (int nt = 0; nt < NTILES; ++nt) {
        int n0 = nt * 16;
        // GEMM1 A fragment: Q rows n0..n0+15 (upper K half zero)
        v16h a = {};
        v16h qa = *(const v16h*)(qbase + (size_t)(n0 + lr) * DDIM);
#pragma unroll
        for (int h = 0; h < 16; ++h) a[h] = g ? (_Float16)0 : qa[h];

        v8f s = {};
        s = __builtin_amdgcn_wmma_f32_16x16x32_f16(false, a, false, bk,
                                                   (short)0, s, false, false);
        // p[r] = exp(S[n0+r+8g][m0+lr] - rowmax[n0+r+8g]); 32B vector rowmax load
        v8f rmv = *(const v8f*)(rmbase + n0 + 8 * g);
        v8h p;
#pragma unroll
        for (int r = 0; r < 8; ++r)
            p[r] = (_Float16)__expf(s[r] - rmv[r]);

        __syncthreads();                          // protect previous read of Pt
        *(v8h*)(&Pt[lr * 16 + 8 * g]) = p;        // transposed store: Pt[m][n]
        __syncthreads();

        // GEMM2 B fragment: column lr of P (contiguous in Pt); lanes 16-31
        // feed K=16..31 against the zero upper half of va -> payload harmless.
        v16h bp = *(const v16h*)(&Pt[lr * 16]);

        // GEMM2 A fragment: Vs[d=lr][n0 .. n0+15] in lane half 0, zero in lane half 1
        v16h va = {};
        v16h vv = *(const v16h*)(vbase + (size_t)lr * NPIX + n0);
#pragma unroll
        for (int h = 0; h < 16; ++h) va[h] = g ? (_Float16)0 : vv[h];

        acc = __builtin_amdgcn_wmma_f32_16x16x32_f16(false, va, false, bp,
                                                     (short)0, acc, false, false);
    }
    // C layout: acc[r] = out[d = r+8g][m0+lr]
#pragma unroll
    for (int r = 0; r < 8; ++r) {
        out[((size_t)b * DDIM + r + 8 * g) * NPIX + m0 + lr] = acc[r];
    }
}

// ---------------------------------------------------------------------------
extern "C" void kernel_launch(void* const* d_in, const int* in_sizes, int n_in,
                              void* d_out, int out_size, void* d_ws, size_t ws_size,
                              hipStream_t stream) {
    const float* x  = (const float*)d_in[0];
    const float* e  = (const float*)d_in[1];
    const float* Wq = (const float*)d_in[2];
    const float* bq = (const float*)d_in[3];
    const float* Wk = (const float*)d_in[4];
    const float* bk = (const float*)d_in[5];
    const float* Wv = (const float*)d_in[6];
    const float* bv = (const float*)d_in[7];
    float* out = (float*)d_out;

    // workspace layout (3.125 MB total)
    char* ws = (char*)d_ws;
    const size_t MB = 1u << 20;
    _Float16* Qt  = (_Float16*)(ws + 0 * MB);      // (B,N,16) f16 : 1 MB
    _Float16* Kt  = (_Float16*)(ws + 1 * MB);      // (B,N,16) f16 : 1 MB
    _Float16* Vh  = (_Float16*)(ws + 2 * MB);      // (B,16,N) f16 : 1 MB
    float*    rmx = (float*)(ws + 3 * MB);         // (B,N) f32 : 128 KB

    (void)in_sizes; (void)n_in; (void)out_size; (void)ws_size;

    // A: QKV projection (Vh rewritten every call -> in-place rescale is safe)
    ea_qkv_kernel<<<(BATCH * NPIX + 255) / 256, 256, 0, stream>>>(
        x, e, Wq, bq, Wk, bk, Wv, bv, Qt, Kt, Vh);

    // B: softmax row stats (max sweep + sum sweep) + in-place V/Z rescale
    ea_rowstats_kernel<<<BATCH * NTILES, 32, 0, stream>>>(Qt, Kt, Vh, rmx);

    // D: attention output (one wave per 16-col m-tile)
    ea_attnout_kernel<<<BATCH * NTILES, 32, 0, stream>>>(Qt, Kt, Vh, rmx, out);
}
